// GcnBlock_3169685865283
// MI455X (gfx1250) — compile-verified, matrix operand order checked by reading, not run on hardware
//
#include <hip/hip_runtime.h>

// ---------------- problem constants (match reference) ----------------
#define NN    50000
#define EE    800000
#define DIN   512
#define DOUT  256

// ---------------- CDNA5 WMMA types ----------------
typedef __attribute__((ext_vector_type(16))) __bf16 v16bf;
typedef __attribute__((ext_vector_type(8)))  float  v8f;

__device__ __forceinline__ unsigned short f2bf(float f) {
  // round-to-nearest-even f32 -> bf16
  unsigned u = __builtin_bit_cast(unsigned, f);
  u = (u + 0x7FFFu + ((u >> 16) & 1u)) >> 16;
  return (unsigned short)u;
}

// ---------------- utility: zero a float buffer ----------------
__global__ void gcn_zero(float* __restrict__ p, int n) {
  int i = blockIdx.x * 256 + threadIdx.x;
  if (i < n) p[i] = 0.0f;
}

// ---------------- degree accumulation (targets = edge_index[1]) ----------------
__global__ void gcn_deg(const int* __restrict__ ei, float* __restrict__ deg) {
  int e = blockIdx.x * 256 + threadIdx.x;
  if (e < EE) {
    int c = ei[EE + e];
    unsafeAtomicAdd(&deg[c], 1.0f);   // native global_atomic_add_f32
  }
}

// ---------------- dinv = deg>0 ? rsqrt(deg) : 0 ----------------
__global__ void gcn_dinv(const float* __restrict__ deg, float* __restrict__ dinv) {
  int i = blockIdx.x * 256 + threadIdx.x;
  if (i < NN) {
    float d = deg[i];
    dinv[i] = (d > 0.0f) ? __frsqrt_rn(d) : 0.0f;
  }
}

// ---------------- GEMM: h[N,256] = x[N,512] @ W[256,512]^T via bf16 WMMA ----------------
// Block = 256 threads = 8 waves. Block M-tile = 128 rows (16 per wave).
// Each wave holds 16 accumulators (full 256-wide output strip).
#define LDSTR 34   // ushort row stride (17 dwords) -> LDS bank-conflict free

__global__ __launch_bounds__(256, 1) void gcn_gemm_wmma(
    const float* __restrict__ x, const float* __restrict__ W,
    float* __restrict__ h) {
  __shared__ unsigned short Wl[DOUT * LDSTR];  // 256 cols x 32 k (bf16), padded
  __shared__ unsigned short Al[128  * LDSTR];  // 128 rows x 32 k (bf16), padded

  const int t    = threadIdx.x;
  const int wave = t >> 5;
  const int lane = t & 31;
  const int hl   = lane >> 4;    // half-lane selects K group
  const int l16  = lane & 15;
  const int m0   = blockIdx.x * 128;

  v8f acc[16];
#pragma unroll
  for (int i = 0; i < 16; ++i) acc[i] = (v8f){};

  for (int k0 = 0; k0 < DIN; k0 += 32) {
    __syncthreads();
    // stage W tile: 256 cols x 32 k, f32 -> bf16 (coalesced: a wave covers one col's 32 k)
#pragma unroll
    for (int i = 0; i < 32; ++i) {
      int idx = i * 256 + t;
      int col = idx >> 5, kk = idx & 31;
      Wl[col * LDSTR + kk] = f2bf(W[col * DIN + k0 + kk]);
    }
    // stage A tile: 128 rows x 32 k
#pragma unroll
    for (int i = 0; i < 16; ++i) {
      int idx = i * 256 + t;
      int row = idx >> 5, kk = idx & 31;
      int r = m0 + row; if (r >= NN) r = NN - 1;   // clamp (stores guarded below)
      Al[row * LDSTR + kk] = f2bf(x[(size_t)r * DIN + k0 + kk]);
    }
    __syncthreads();

    // A fragment: 16-bit A 16x32 layout (lanes 0-15 M, VGPR v holds K pairs; half-lane +8/+24)
    union { unsigned u[8]; v16bf v; } afrag;
    {
      const unsigned* Arow = (const unsigned*)(Al + ((wave << 4) + l16) * LDSTR);
#pragma unroll
      for (int v = 0; v < 8; ++v) {
        int dw = (v < 4) ? (hl * 4 + v) : (8 + hl * 4 + (v - 4));
        afrag.u[v] = Arow[dw];
      }
    }

    // 16 N-tiles: B fragment mirrors A layout (W rows are k-contiguous = B^T rows)
#pragma unroll
    for (int nt = 0; nt < 16; ++nt) {
      union { unsigned u[8]; v16bf v; } bfrag;
      const unsigned* Brow = (const unsigned*)(Wl + ((nt << 4) + l16) * LDSTR);
#pragma unroll
      for (int v = 0; v < 8; ++v) {
        int dw = (v < 4) ? (hl * 4 + v) : (8 + hl * 4 + (v - 4));
        bfrag.u[v] = Brow[dw];
      }
      acc[nt] = __builtin_amdgcn_wmma_f32_16x16x32_bf16(
          false, afrag.v, false, bfrag.v, (short)0, acc[nt], false, false);
    }
  }

  // store: C/D layout -> lanes 0-15: M=j, N=lane; lanes 16-31: M=j+8, N=lane-16
#pragma unroll
  for (int nt = 0; nt < 16; ++nt) {
#pragma unroll
    for (int j = 0; j < 8; ++j) {
      int row = m0 + (wave << 4) + hl * 8 + j;
      if (row < NN) h[(size_t)row * DOUT + (nt << 4) + l16] = acc[nt][j];
    }
  }
}

// ---------------- edge scatter: out[col] += h[row] * dinv[row]*dinv[col] ----------------
// 64 threads per edge, float4 per thread; h and out live in 192MB L2.
__global__ void gcn_scatter(const float* __restrict__ h, const int* __restrict__ ei,
                            const float* __restrict__ dinv, float* __restrict__ out) {
  long long idx = (long long)blockIdx.x * 256 + threadIdx.x;
  int e  = (int)(idx >> 6);
  int c4 = ((int)idx & 63) << 2;
  if (e >= EE) return;
  int src = ei[e];         // row (source)
  int dst = ei[EE + e];    // col (target)
  float nrm = dinv[src] * dinv[dst];
  const float4 hv = *(const float4*)(h + (size_t)src * DOUT + c4);
  float* o = out + (size_t)dst * DOUT + c4;
  unsafeAtomicAdd(o + 0, hv.x * nrm);
  unsafeAtomicAdd(o + 1, hv.y * nrm);
  unsafeAtomicAdd(o + 2, hv.z * nrm);
  unsafeAtomicAdd(o + 3, hv.w * nrm);
}

// ---------------- finalize: out = relu(out + b) in place ----------------
__global__ void gcn_finalize(float* __restrict__ out, const float* __restrict__ b) {
  long long i = (long long)blockIdx.x * 256 + threadIdx.x;
  if (i < (long long)NN * DOUT) {
    float v = out[i] + b[(int)(i % DOUT)];
    out[i] = fmaxf(v, 0.0f);
  }
}

// ---------------- launcher ----------------
extern "C" void kernel_launch(void* const* d_in, const int* in_sizes, int n_in,
                              void* d_out, int out_size, void* d_ws, size_t ws_size,
                              hipStream_t stream) {
  const float* x  = (const float*)d_in[0];
  const float* W  = (const float*)d_in[1];
  const float* b  = (const float*)d_in[2];
  const int*   ei = (const int*)d_in[3];
  float* out = (float*)d_out;

  float* h    = (float*)d_ws;                       // N*DOUT f32 = 51.2 MB
  float* deg  = h + (size_t)NN * DOUT;              // N f32
  float* dinv = deg + NN;                           // N f32

  // zero output accumulator and degree buffer (every call: deterministic)
  gcn_zero<<<(NN * DOUT + 255) / 256, 256, 0, stream>>>(out, NN * DOUT);
  gcn_zero<<<(NN + 255) / 256, 256, 0, stream>>>(deg, NN);

  gcn_deg<<<(EE + 255) / 256, 256, 0, stream>>>(ei, deg);
  gcn_dinv<<<(NN + 255) / 256, 256, 0, stream>>>(deg, dinv);

  gcn_gemm_wmma<<<(NN + 127) / 128, 256, 0, stream>>>(x, W, h);

  long long sitems = (long long)EE * 64;
  gcn_scatter<<<(unsigned)((sitems + 255) / 256), 256, 0, stream>>>(h, ei, dinv, out);

  long long fitems = (long long)NN * DOUT;
  gcn_finalize<<<(unsigned)((fitems + 255) / 256), 256, 0, stream>>>(out, b);
}
